// MOOZYSlideEncoder_34144990003455
// MI455X (gfx1250) — matrix operand order repaired and checked by reading, not compiled, not run on hardware
//
#include <hip/hip_runtime.h>
#include <hip/hip_bf16.h>
#include <math.h>

typedef __attribute__((ext_vector_type(16))) _Float16 v16h;
typedef __attribute__((ext_vector_type(8)))  _Float16 v8h;
typedef __attribute__((ext_vector_type(8)))  float    v8f;

union Frag { v16h v; v8h h8[2]; };

#define BB    2
#define NTOK  1024
#define RREG  4
#define SS    1029      /* 1 + R + N */
#define DD    768
#define HH    12
#define HDIM  64
#define FFD   3072
#define LL    4
#define FEATD 384
#define SKP   1056      /* keys padded to multiple of 32 */

__device__ __forceinline__ float geluf(float x) {
  return 0.5f * x * (1.0f + erff(x * 0.70710678118654752f));
}

// ---------------- elementwise helpers ----------------
__global__ void f32_to_f16_kernel(const float* __restrict__ in, _Float16* __restrict__ out, int n) {
  int i = blockIdx.x * blockDim.x + threadIdx.x;
  if (i < n) out[i] = (_Float16)in[i];
}

// Wt[n*K + k] = W[k*N + n]   (transpose + fp32->fp16)
__global__ void wconv_kernel(const float* __restrict__ W, _Float16* __restrict__ Wt, int K, int N) {
  int i = blockIdx.x * blockDim.x + threadIdx.x;
  if (i >= K * N) return;
  int n = i / K, k = i % K;
  Wt[i] = (_Float16)W[(size_t)k * N + n];
}

__global__ void tok_init_kernel(float* __restrict__ h, const float* __restrict__ cls,
                                const float* __restrict__ reg) {
  int i = blockIdx.x * blockDim.x + threadIdx.x;
  int tot = BB * (1 + RREG) * DD;
  if (i >= tot) return;
  int d = i % DD; int rest = i / DD;
  int s = rest % (1 + RREG); int b = rest / (1 + RREG);
  float v = (s == 0) ? cls[d] : reg[(s - 1) * DD + d];
  h[((size_t)b * SS + s) * DD + d] = v;
}

__global__ void mask_apply_kernel(float* __restrict__ h, const unsigned char* __restrict__ mask,
                                  const float* __restrict__ mtok) {
  int i = blockIdx.x * blockDim.x + threadIdx.x;
  if (i >= BB * NTOK * DD) return;
  int d = i % DD; int rest = i / DD;
  int n = rest % NTOK; int b = rest / NTOK;
  if (mask[b * NTOK + n]) h[((size_t)b * SS + (1 + RREG) + n) * DD + d] = mtok[d];
}

__global__ void mask_out_kernel(const unsigned char* __restrict__ mask, float* __restrict__ out) {
  int i = blockIdx.x * blockDim.x + threadIdx.x;
  if (i < BB * NTOK) out[i] = mask[i] ? 1.0f : 0.0f;
}

// v32 (B,S,D) -> vT16 (B,H,HD,SKP), zero padded keys
__global__ void v_pack_kernel(const float* __restrict__ v32, _Float16* __restrict__ vT) {
  size_t i = (size_t)blockIdx.x * blockDim.x + threadIdx.x;
  size_t tot = (size_t)BB * HH * HDIM * SKP;
  if (i >= tot) return;
  int s = (int)(i % SKP); size_t r = i / SKP;
  int d = (int)(r % HDIM); r /= HDIM;
  int hh = (int)(r % HH);  int b = (int)(r / HH);
  vT[i] = (s < SS) ? (_Float16)v32[((size_t)b * SS + s) * DD + hh * HDIM + d] : (_Float16)0.0f;
}

// ---------------- layernorm kernels ----------------
__global__ __launch_bounds__(256) void ln_rows_f16_kernel(const float* __restrict__ X,
    const float* __restrict__ g, const float* __restrict__ b, _Float16* __restrict__ Y, int D) {
  __shared__ float red[256];
  size_t row = blockIdx.x;
  const float* x = X + row * D;
  float s = 0.f;
  for (int i = threadIdx.x; i < D; i += 256) s += x[i];
  red[threadIdx.x] = s; __syncthreads();
  for (int k = 128; k > 0; k >>= 1) { if (threadIdx.x < k) red[threadIdx.x] += red[threadIdx.x + k]; __syncthreads(); }
  float mean = red[0] / D; __syncthreads();
  float vs = 0.f;
  for (int i = threadIdx.x; i < D; i += 256) { float d = x[i] - mean; vs += d * d; }
  red[threadIdx.x] = vs; __syncthreads();
  for (int k = 128; k > 0; k >>= 1) { if (threadIdx.x < k) red[threadIdx.x] += red[threadIdx.x + k]; __syncthreads(); }
  float rstd = rsqrtf(red[0] / D + 1e-5f);
  for (int i = threadIdx.x; i < D; i += 256)
    Y[row * D + i] = (_Float16)((x[i] - mean) * rstd * g[i] + b[i]);
}

__global__ __launch_bounds__(256) void ln_final_kernel(const float* __restrict__ X,
    const float* __restrict__ g, const float* __restrict__ b, float* __restrict__ out) {
  __shared__ float red[256];
  size_t row = blockIdx.x;
  int bb = (int)(row / SS), s = (int)(row % SS);
  const float* xr = X + row * DD;
  float sm = 0.f;
  for (int i = threadIdx.x; i < DD; i += 256) sm += xr[i];
  red[threadIdx.x] = sm; __syncthreads();
  for (int k = 128; k > 0; k >>= 1) { if (threadIdx.x < k) red[threadIdx.x] += red[threadIdx.x + k]; __syncthreads(); }
  float mean = red[0] / DD; __syncthreads();
  float vs = 0.f;
  for (int i = threadIdx.x; i < DD; i += 256) { float d = xr[i] - mean; vs += d * d; }
  red[threadIdx.x] = vs; __syncthreads();
  for (int k = 128; k > 0; k >>= 1) { if (threadIdx.x < k) red[threadIdx.x] += red[threadIdx.x + k]; __syncthreads(); }
  float rstd = rsqrtf(red[0] / DD + 1e-5f);
  float* dst = nullptr;
  if (s == 0) dst = out + (size_t)bb * DD;
  else if (s >= 1 + RREG) dst = out + (size_t)BB * DD + ((size_t)bb * NTOK + (s - (1 + RREG))) * DD;
  if (dst)
    for (int i = threadIdx.x; i < DD; i += 256) dst[i] = (xr[i] - mean) * rstd * g[i] + b[i];
}

// per-head LN of q and k over HD=64, pack to (B,H,S,HD) fp16; one wave per (b,s,head)
__global__ __launch_bounds__(256) void qk_ln_pack_kernel(
    const float* __restrict__ q32, const float* __restrict__ k32,
    const float* __restrict__ qg, const float* __restrict__ qb,
    const float* __restrict__ kg, const float* __restrict__ kbta,
    _Float16* __restrict__ q16, _Float16* __restrict__ k16) {
  int wg = (blockIdx.x * 256 + threadIdx.x) >> 5;
  int lane = threadIdx.x & 31;
  int total = BB * SS * HH;
  if (wg >= total) return;
  int b = wg / (SS * HH); int rem = wg % (SS * HH);
  int s = rem / HH; int hh = rem % HH;
  size_t srcBase = ((size_t)b * SS + s) * DD + hh * HDIM;
  size_t dstBase = (((size_t)b * HH + hh) * SS + s) * HDIM;
  int d0 = lane * 2;
  {
    float v0 = q32[srcBase + d0], v1 = q32[srcBase + d0 + 1];
    float sum = v0 + v1;
    for (int o = 1; o < 32; o <<= 1) sum += __shfl_xor(sum, o, 32);
    float mean = sum * (1.0f / HDIM);
    float a0 = v0 - mean, a1 = v1 - mean;
    float vv = a0 * a0 + a1 * a1;
    for (int o = 1; o < 32; o <<= 1) vv += __shfl_xor(vv, o, 32);
    float rstd = rsqrtf(vv * (1.0f / HDIM) + 1e-5f);
    q16[dstBase + d0]     = (_Float16)(a0 * rstd * qg[d0] + qb[d0]);
    q16[dstBase + d0 + 1] = (_Float16)(a1 * rstd * qg[d0 + 1] + qb[d0 + 1]);
  }
  {
    float v0 = k32[srcBase + d0], v1 = k32[srcBase + d0 + 1];
    float sum = v0 + v1;
    for (int o = 1; o < 32; o <<= 1) sum += __shfl_xor(sum, o, 32);
    float mean = sum * (1.0f / HDIM);
    float a0 = v0 - mean, a1 = v1 - mean;
    float vv = a0 * a0 + a1 * a1;
    for (int o = 1; o < 32; o <<= 1) vv += __shfl_xor(vv, o, 32);
    float rstd = rsqrtf(vv * (1.0f / HDIM) + 1e-5f);
    k16[dstBase + d0]     = (_Float16)(a0 * rstd * kg[d0] + kbta[d0]);
    k16[dstBase + d0 + 1] = (_Float16)(a1 * rstd * kg[d0 + 1] + kbta[d0 + 1]);
  }
}

// ---------------- WMMA GEMM: C = A(MxK) * Bt^T(KxN) + bias ----------------
// Wave tile 32(M) x 64(N): 8 WMMAs per 12 b128 loads. Block = 4 waves -> 64x128 tile.
// EPI 0: store f32; 1: gelu -> f16; 2: f32 residual accumulate; 3: gelu -> f32 with embed row remap
template<int EPI>
__global__ __launch_bounds__(128) void wgemm_kernel(
    const _Float16* __restrict__ A,   // M x K row-major
    const _Float16* __restrict__ Bt,  // N x K row-major (pre-transposed weight)
    const float* __restrict__ bias,   // N
    float* __restrict__ Cf, _Float16* __restrict__ Ch,
    int M, int N, int K) {
  int w = threadIdx.x >> 5;
  int lane = threadIdx.x & 31;
  int lm = lane & 15, half = lane >> 4;
  int m0 = blockIdx.x * 64 + (w & 1) * 32;
  int n0 = blockIdx.y * 128 + (w >> 1) * 64;
  int ra0 = m0 + lm;      if (ra0 > M - 1) ra0 = M - 1;
  int ra1 = m0 + 16 + lm; if (ra1 > M - 1) ra1 = M - 1;
  const _Float16* pa0 = A + (size_t)ra0 * K;
  const _Float16* pa1 = A + (size_t)ra1 * K;
  const _Float16* pb0 = Bt + (size_t)(n0 + lm) * K;
  const _Float16* pb1 = Bt + (size_t)(n0 + 16 + lm) * K;
  const _Float16* pb2 = Bt + (size_t)(n0 + 32 + lm) * K;
  const _Float16* pb3 = Bt + (size_t)(n0 + 48 + lm) * K;
  v8f c00 = {}, c01 = {}, c02 = {}, c03 = {};
  v8f c10 = {}, c11 = {}, c12 = {}, c13 = {};
  for (int k0 = 0; k0 < K; k0 += 32) {
    Frag a0, a1, b0, b1, b2, b3;
    a0.h8[0] = *(const v8h*)(pa0 + k0 + half * 16);
    a0.h8[1] = *(const v8h*)(pa0 + k0 + half * 16 + 8);
    a1.h8[0] = *(const v8h*)(pa1 + k0 + half * 16);
    a1.h8[1] = *(const v8h*)(pa1 + k0 + half * 16 + 8);
    b0.h8[0] = *(const v8h*)(pb0 + k0 + half * 16);
    b0.h8[1] = *(const v8h*)(pb0 + k0 + half * 16 + 8);
    b1.h8[0] = *(const v8h*)(pb1 + k0 + half * 16);
    b1.h8[1] = *(const v8h*)(pb1 + k0 + half * 16 + 8);
    b2.h8[0] = *(const v8h*)(pb2 + k0 + half * 16);
    b2.h8[1] = *(const v8h*)(pb2 + k0 + half * 16 + 8);
    b3.h8[0] = *(const v8h*)(pb3 + k0 + half * 16);
    b3.h8[1] = *(const v8h*)(pb3 + k0 + half * 16 + 8);
    if (k0 + 32 < K) {  // gfx1250 global_prefetch of next K slice (L2-resident stream)
      __builtin_prefetch(pa0 + k0 + 32, 0, 3);
      __builtin_prefetch(pa1 + k0 + 32, 0, 3);
      __builtin_prefetch(pb0 + k0 + 32, 0, 3);
      __builtin_prefetch(pb2 + k0 + 32, 0, 3);
    }
    c00 = __builtin_amdgcn_wmma_f32_16x16x32_f16(false, a0.v, false, b0.v, (short)0, c00, false, false);
    c01 = __builtin_amdgcn_wmma_f32_16x16x32_f16(false, a0.v, false, b1.v, (short)0, c01, false, false);
    c02 = __builtin_amdgcn_wmma_f32_16x16x32_f16(false, a0.v, false, b2.v, (short)0, c02, false, false);
    c03 = __builtin_amdgcn_wmma_f32_16x16x32_f16(false, a0.v, false, b3.v, (short)0, c03, false, false);
    c10 = __builtin_amdgcn_wmma_f32_16x16x32_f16(false, a1.v, false, b0.v, (short)0, c10, false, false);
    c11 = __builtin_amdgcn_wmma_f32_16x16x32_f16(false, a1.v, false, b1.v, (short)0, c11, false, false);
    c12 = __builtin_amdgcn_wmma_f32_16x16x32_f16(false, a1.v, false, b2.v, (short)0, c12, false, false);
    c13 = __builtin_amdgcn_wmma_f32_16x16x32_f16(false, a1.v, false, b3.v, (short)0, c13, false, false);
  }
  int ncol0 = n0 + lm, ncol1 = n0 + 16 + lm, ncol2 = n0 + 32 + lm, ncol3 = n0 + 48 + lm;
  float bias0 = bias[ncol0], bias1 = bias[ncol1], bias2 = bias[ncol2], bias3 = bias[ncol3];
  auto emit = [&](int m, int n, float e) {
    if (EPI == 0)      Cf[(size_t)m * N + n] = e;
    else if (EPI == 1) Ch[(size_t)m * N + n] = (_Float16)geluf(e);
    else if (EPI == 2) Cf[(size_t)m * N + n] += e;
    else { size_t row = (size_t)(m / NTOK) * SS + (1 + RREG) + (m % NTOK);
           Cf[row * N + n] = geluf(e); }
  };
  #pragma unroll
  for (int v = 0; v < 8; ++v) {
    int mr0 = m0 + v + 8 * half;
    int mr1 = mr0 + 16;
    if (mr0 < M) {
      emit(mr0, ncol0, c00[v] + bias0); emit(mr0, ncol1, c01[v] + bias1);
      emit(mr0, ncol2, c02[v] + bias2); emit(mr0, ncol3, c03[v] + bias3);
    }
    if (mr1 < M) {
      emit(mr1, ncol0, c10[v] + bias0); emit(mr1, ncol1, c11[v] + bias1);
      emit(mr1, ncol2, c12[v] + bias2); emit(mr1, ncol3, c13[v] + bias3);
    }
  }
}

// ---------------- flash attention: one wave per 16-query tile ----------------
__global__ __launch_bounds__(128) void attn_kernel(
    const _Float16* __restrict__ q16,   // (B,H,S,64)
    const _Float16* __restrict__ k16,   // (B,H,S,64)
    const _Float16* __restrict__ vT16,  // (B,H,64,SKP)
    const float* __restrict__ coords,   // (B,N,2)
    const float* __restrict__ psz,      // (B)
    _Float16* __restrict__ o16) {       // (B,S,D) f16
  __shared__ _Float16 plds[4][16 * 32];
  int wid = threadIdx.x >> 5;
  int lane = threadIdx.x & 31;
  int lm = lane & 15, half = lane >> 4;
  int bh = blockIdx.y;
  int b = bh / HH, hh = bh % HH;
  int nQT = (SS + 15) / 16;
  int qt = blockIdx.x * 4 + wid;
  if (qt >= nQT) return;   // wave-uniform; EXEC stays all-ones for active waves

  float slope = (hh < 8) ? -exp2f(-(float)(hh + 1)) : -exp2f(-0.5f - (float)(hh - 8));
  float ips = 1.0f / fmaxf(psz[b], 1e-6f);
  const float scl = 0.125f;   // 1/sqrt(64)

  int mq = qt * 16 + lm; if (mq > SS - 1) mq = SS - 1;
  const _Float16* qrow = q16 + (((size_t)(b * HH + hh) * SS) + mq) * HDIM;
  Frag aq0, aq1;
  aq0.h8[0] = *(const v8h*)(qrow + half * 16);
  aq0.h8[1] = *(const v8h*)(qrow + half * 16 + 8);
  aq1.h8[0] = *(const v8h*)(qrow + 32 + half * 16);
  aq1.h8[1] = *(const v8h*)(qrow + 32 + half * 16 + 8);

  float qx[8], qy[8]; bool qp[8];
  #pragma unroll
  for (int v = 0; v < 8; ++v) {
    int m = qt * 16 + v + 8 * half;
    bool p = (m >= 1 + RREG) && (m < SS);
    qp[v] = p;
    int pi = p ? (m - (1 + RREG)) : 0;
    qx[v] = coords[((size_t)b * NTOK + pi) * 2 + 0];
    qy[v] = coords[((size_t)b * NTOK + pi) * 2 + 1];
  }
  float rm[8], rl[8];
  #pragma unroll
  for (int v = 0; v < 8; ++v) { rm[v] = -3.0e38f; rl[v] = 0.f; }
  v8f oc0 = {}, oc1 = {}, oc2 = {}, oc3 = {};

  const _Float16* kbase = k16 + (size_t)(b * HH + hh) * SS * HDIM;
  const _Float16* vbase = vT16 + (size_t)(b * HH + hh) * HDIM * SKP;
  _Float16* pl = &plds[wid][0];

  for (int kb = 0; kb < SKP / 32; ++kb) {
    int j0 = kb * 32 + lm, j1 = j0 + 16;
    int jc0 = (j0 > SS - 1) ? SS - 1 : j0;
    int jc1 = (j1 > SS - 1) ? SS - 1 : j1;
    const _Float16* kr0 = kbase + (size_t)jc0 * HDIM;
    const _Float16* kr1 = kbase + (size_t)jc1 * HDIM;
    // prefetch next key block rows (gfx1250 global_prefetch)
    {
      int jn = kb * 32 + 32 + lm; if (jn > SS - 1) jn = SS - 1;
      __builtin_prefetch(kbase + (size_t)jn * HDIM, 0, 3);
      __builtin_prefetch(vbase + (size_t)lm * SKP + (kb * 32 + 32 < SKP ? kb * 32 + 32 : kb * 32), 0, 3);
    }
    Frag b00, b01, b10, b11;
    b00.h8[0] = *(const v8h*)(kr0 + half * 16);      b00.h8[1] = *(const v8h*)(kr0 + half * 16 + 8);
    b01.h8[0] = *(const v8h*)(kr0 + 32 + half * 16); b01.h8[1] = *(const v8h*)(kr0 + 32 + half * 16 + 8);
    b10.h8[0] = *(const v8h*)(kr1 + half * 16);      b10.h8[1] = *(const v8h*)(kr1 + half * 16 + 8);
    b11.h8[0] = *(const v8h*)(kr1 + 32 + half * 16); b11.h8[1] = *(const v8h*)(kr1 + 32 + half * 16 + 8);
    v8f s0 = {}, s1 = {};
    s0 = __builtin_amdgcn_wmma_f32_16x16x32_f16(false, aq0.v, false, b00.v, (short)0, s0, false, false);
    s0 = __builtin_amdgcn_wmma_f32_16x16x32_f16(false, aq1.v, false, b01.v, (short)0, s0, false, false);
    s1 = __builtin_amdgcn_wmma_f32_16x16x32_f16(false, aq0.v, false, b10.v, (short)0, s1, false, false);
    s1 = __builtin_amdgcn_wmma_f32_16x16x32_f16(false, aq1.v, false, b11.v, (short)0, s1, false, false);

    bool kp0 = (j0 >= 1 + RREG) && (j0 < SS);
    bool kp1 = (j1 >= 1 + RREG) && (j1 < SS);
    int pj0 = kp0 ? (j0 - (1 + RREG)) : 0;
    int pj1 = kp1 ? (j1 - (1 + RREG)) : 0;
    float kx0 = coords[((size_t)b * NTOK + pj0) * 2 + 0];
    float ky0 = coords[((size_t)b * NTOK + pj0) * 2 + 1];
    float kx1 = coords[((size_t)b * NTOK + pj1) * 2 + 0];
    float ky1 = coords[((size_t)b * NTOK + pj1) * 2 + 1];
    bool jv0 = (j0 < SS), jv1 = (j1 < SS);

    #pragma unroll
    for (int v = 0; v < 8; ++v) {
      float dx0 = qx[v] - kx0, dy0 = qy[v] - ky0;
      float dx1 = qx[v] - kx1, dy1 = qy[v] - ky1;
      float bias0 = (qp[v] && kp0) ? slope * sqrtf(dx0 * dx0 + dy0 * dy0) * ips : 0.f;
      float bias1 = (qp[v] && kp1) ? slope * sqrtf(dx1 * dx1 + dy1 * dy1) * ips : 0.f;
      float e0 = jv0 ? (s0[v] * scl + bias0) : -1.0e30f;
      float e1 = jv1 ? (s1[v] * scl + bias1) : -1.0e30f;
      float mx = fmaxf(e0, e1);
      #pragma unroll
      for (int o = 1; o < 16; o <<= 1) mx = fmaxf(mx, __shfl_xor(mx, o, 32));
      float mnew = fmaxf(rm[v], mx);
      float corr = __expf(rm[v] - mnew);
      float p0 = __expf(e0 - mnew);
      float p1 = __expf(e1 - mnew);
      float sum = p0 + p1;
      #pragma unroll
      for (int o = 1; o < 16; o <<= 1) sum += __shfl_xor(sum, o, 32);
      rl[v] = rl[v] * corr + sum;
      rm[v] = mnew;
      oc0[v] *= corr; oc1[v] *= corr; oc2[v] *= corr; oc3[v] *= corr;
      int r = v + 8 * half;
      pl[r * 32 + lm]      = (_Float16)p0;
      pl[r * 32 + 16 + lm] = (_Float16)p1;
    }
    // re-read P in A-fragment layout (intra-wave LDS dependency -> dscnt waits)
    Frag ap;
    ap.h8[0] = *(const v8h*)(pl + lm * 32 + half * 16);
    ap.h8[1] = *(const v8h*)(pl + lm * 32 + half * 16 + 8);
    const _Float16* vr = vbase + (size_t)lm * SKP + kb * 32;
    Frag bvf;
    bvf.h8[0] = *(const v8h*)(vr + half * 16); bvf.h8[1] = *(const v8h*)(vr + half * 16 + 8);
    oc0 = __builtin_amdgcn_wmma_f32_16x16x32_f16(false, ap.v, false, bvf.v, (short)0, oc0, false, false);
    vr += (size_t)16 * SKP;
    bvf.h8[0] = *(const v8h*)(vr + half * 16); bvf.h8[1] = *(const v8h*)(vr + half * 16 + 8);
    oc1 = __builtin_amdgcn_wmma_f32_16x16x32_f16(false, ap.v, false, bvf.v, (short)0, oc1, false, false);
    vr += (size_t)16 * SKP;
    bvf.h8[0] = *(const v8h*)(vr + half * 16); bvf.h8[1] = *(const v8h*)(vr + half * 16 + 8);
    oc2 = __builtin_amdgcn_wmma_f32_16x16x32_f16(false, ap.v, false, bvf.v, (short)0, oc2, false, false);
    vr += (size_t)16 * SKP;
    bvf.h8[0] = *(const v8h*)(vr + half * 16); bvf.h8[1] = *(const v8h*)(vr + half * 16 + 8);
    oc3 = __builtin_amdgcn_wmma_f32_16x16x32_f16(false, ap.v, false, bvf.v, (short)0, oc3, false, false);
  }
  #pragma unroll
  for (int v = 0; v < 8; ++v) {
    int m = qt * 16 + v + 8 * half;
    if (m < SS) {
      float inv = (rl[v] > 0.f) ? 1.0f / rl[v] : 0.f;
      size_t base = ((size_t)b * SS + m) * DD + hh * HDIM;
      o16[base + 0 * 16 + lm] = (_Float16)(oc0[v] * inv);
      o16[base + 1 * 16 + lm] = (_Float16)(oc1[v] * inv);
      o16[base + 2 * 16 + lm] = (_Float16)(oc2[v] * inv);
      o16[base + 3 * 16 + lm] = (_Float16)(oc3[v] * inv);
    }
  }
}

// ---------------- host side ----------------
extern "C" void kernel_launch(void* const* d_in, const int* in_sizes, int n_in,
                              void* d_out, int out_size, void* d_ws, size_t ws_size,
                              hipStream_t stream) {
  (void)in_sizes; (void)n_in; (void)out_size; (void)ws_size;
  const float*         x      = (const float*)d_in[0];
  const unsigned char* mask   = (const unsigned char*)d_in[1];
  const float*         coords = (const float*)d_in[2];
  const float*         psz    = (const float*)d_in[3];
  const float*         Win    = (const float*)d_in[4];
  const float*         bin_   = (const float*)d_in[5];
  const float*         cls    = (const float*)d_in[6];
  const float*         reg    = (const float*)d_in[7];
  const float*         mtok   = (const float*)d_in[8];
  const float*         ln1g   = (const float*)d_in[9];
  const float*         ln1b   = (const float*)d_in[10];
  const float*         Wq     = (const float*)d_in[11];
  const float*         bq     = (const float*)d_in[12];
  const float*         Wk     = (const float*)d_in[13];
  const float*         bk     = (const float*)d_in[14];
  const float*         Wv     = (const float*)d_in[15];
  const float*         bvv    = (const float*)d_in[16];
  const float*         qng    = (const float*)d_in[17];
  const float*         qnb    = (const float*)d_in[18];
  const float*         kng    = (const float*)d_in[19];
  const float*         knb    = (const float*)d_in[20];
  const float*         Wo     = (const float*)d_in[21];
  const float*         bo     = (const float*)d_in[22];
  const float*         ln2g   = (const float*)d_in[23];
  const float*         ln2b   = (const float*)d_in[24];
  const float*         W1     = (const float*)d_in[25];
  const float*         b1     = (const float*)d_in[26];
  const float*         W2     = (const float*)d_in[27];
  const float*         b2     = (const float*)d_in[28];
  const float*         lnfg   = (const float*)d_in[29];
  const float*         lnfb   = (const float*)d_in[30];
  float* out = (float*)d_out;

  char* ws = (char*)d_ws;
  size_t off = 0;
  auto carve = [&](size_t bytes) { char* p = ws + off; off += ((bytes + 255) & ~(size_t)255); return p; };
  const int BS = BB * SS;  // 2058
  float*    h    = (float*)   carve((size_t)BS * DD * 4);
  _Float16* xn16 = (_Float16*)carve((size_t)BS * DD * 2);
  _Float16* x16  = (_Float16*)carve((size_t)BB * NTOK * FEATD * 2);
  float*    q32  = (float*)   carve((size_t)BS * DD * 4);
  float*    k32  = (float*)   carve((size_t)BS * DD * 4);
  float*    v32  = (float*)   carve((size_t)BS * DD * 4);
  _Float16* q16  = (_Float16*)carve((size_t)BS * DD * 2);
  _Float16* k16  = (_Float16*)carve((size_t)BS * DD * 2);
  _Float16* vT16 = (_Float16*)carve((size_t)BB * HH * HDIM * SKP * 2);
  _Float16* o16  = (_Float16*)carve((size_t)BS * DD * 2);
  _Float16* ff16 = (_Float16*)carve((size_t)BS * FFD * 2);
  _Float16* Wint = (_Float16*)carve((size_t)DD * FEATD * 2);
  _Float16* wbuf = (_Float16*)carve((size_t)(4 * DD * DD + 2 * DD * FFD) * 2);
  _Float16* Wqt = wbuf;
  _Float16* Wkt = wbuf + (size_t)DD * DD;
  _Float16* Wvt = wbuf + (size_t)2 * DD * DD;
  _Float16* Wot = wbuf + (size_t)3 * DD * DD;
  _Float16* W1t = wbuf + (size_t)4 * DD * DD;
  _Float16* W2t = W1t + (size_t)FFD * DD;

  // input conversion + token assembly
  int nx = BB * NTOK * FEATD;
  f32_to_f16_kernel<<<(nx + 255) / 256, 256, 0, stream>>>(x, x16, nx);
  wconv_kernel<<<(DD * FEATD + 255) / 256, 256, 0, stream>>>(Win, Wint, FEATD, DD);
  tok_init_kernel<<<(BB * (1 + RREG) * DD + 255) / 256, 256, 0, stream>>>(h, cls, reg);
  wgemm_kernel<3><<<dim3((BB * NTOK + 63) / 64, DD / 128), 128, 0, stream>>>(
      x16, Wint, bin_, h, nullptr, BB * NTOK, DD, FEATD);
  mask_apply_kernel<<<(BB * NTOK * DD + 255) / 256, 256, 0, stream>>>(h, mask, mtok);

  dim3 gq((BS + 63) / 64, DD / 128);
  for (int l = 0; l < LL; ++l) {
    wconv_kernel<<<(DD * DD + 255) / 256, 256, 0, stream>>>(Wq + (size_t)l * DD * DD, Wqt, DD, DD);
    wconv_kernel<<<(DD * DD + 255) / 256, 256, 0, stream>>>(Wk + (size_t)l * DD * DD, Wkt, DD, DD);
    wconv_kernel<<<(DD * DD + 255) / 256, 256, 0, stream>>>(Wv + (size_t)l * DD * DD, Wvt, DD, DD);
    wconv_kernel<<<(DD * DD + 255) / 256, 256, 0, stream>>>(Wo + (size_t)l * DD * DD, Wot, DD, DD);
    wconv_kernel<<<(DD * FFD + 255) / 256, 256, 0, stream>>>(W1 + (size_t)l * DD * FFD, W1t, DD, FFD);
    wconv_kernel<<<(FFD * DD + 255) / 256, 256, 0, stream>>>(W2 + (size_t)l * FFD * DD, W2t, FFD, DD);

    ln_rows_f16_kernel<<<BS, 256, 0, stream>>>(h, ln1g + l * DD, ln1b + l * DD, xn16, DD);
    wgemm_kernel<0><<<gq, 128, 0, stream>>>(xn16, Wqt, bq + l * DD, q32, nullptr, BS, DD, DD);
    wgemm_kernel<0><<<gq, 128, 0, stream>>>(xn16, Wkt, bk + l * DD, k32, nullptr, BS, DD, DD);
    wgemm_kernel<0><<<gq, 128, 0, stream>>>(xn16, Wvt, bvv + l * DD, v32, nullptr, BS, DD, DD);

    int nwaves = BB * SS * HH;
    qk_ln_pack_kernel<<<(nwaves * 32 + 255) / 256, 256, 0, stream>>>(
        q32, k32, qng + l * HDIM, qnb + l * HDIM, kng + l * HDIM, knb + l * HDIM, q16, k16);
    size_t nvp = (size_t)BB * HH * HDIM * SKP;
    v_pack_kernel<<<(unsigned)((nvp + 255) / 256), 256, 0, stream>>>(v32, vT16);

    int nQT = (SS + 15) / 16;
    attn_kernel<<<dim3((nQT + 3) / 4, BB * HH), 128, 0, stream>>>(q16, k16, vT16, coords, psz, o16);

    wgemm_kernel<2><<<gq, 128, 0, stream>>>(o16, Wot, bo + l * DD, h, nullptr, BS, DD, DD);
    ln_rows_f16_kernel<<<BS, 256, 0, stream>>>(h, ln2g + l * DD, ln2b + l * DD, xn16, DD);
    wgemm_kernel<1><<<dim3((BS + 63) / 64, FFD / 128), 128, 0, stream>>>(
        xn16, W1t, b1 + l * FFD, nullptr, ff16, BS, FFD, DD);
    wgemm_kernel<2><<<gq, 128, 0, stream>>>(ff16, W2t, b2 + l * DD, h, nullptr, BS, DD, FFD);
  }

  ln_final_kernel<<<BS, 256, 0, stream>>>(h, lnfg, lnfb, out);
  mask_out_kernel<<<(BB * NTOK + 255) / 256, 256, 0, stream>>>(
      mask, out + (size_t)BB * DD + (size_t)BB * NTOK * DD);
}
